// GCN_eva_19224273617407
// MI455X (gfx1250) — matrix operand order repaired and checked by reading, not compiled, not run on hardware
//
#include <hip/hip_runtime.h>

#define NN 100000
#define NE 3200000
#define F0 512
#define F1 256
#define NC 40

typedef __attribute__((ext_vector_type(16))) __bf16 v16bf;
typedef __attribute__((ext_vector_type(8)))  float  v8f;

__device__ __forceinline__ unsigned bf16pack(float a, float b) {
  // round-to-nearest-even fp32 -> bf16, packed (a in [15:0], b in [31:16])
  unsigned ua = __float_as_uint(a);
  unsigned ub = __float_as_uint(b);
  unsigned ra = (ua + 0x7FFFu + ((ua >> 16) & 1u)) >> 16;
  unsigned rb = (ub + 0x7FFFu + ((ub >> 16) & 1u)) >> 16;
  return (rb << 16) | (ra & 0xFFFFu);
}

// ---------------------------------------------------------------------------
// Pack a row-major fp32 weight matrix W[Ktot][Ntot] into bf16 WMMA B-fragment
// order: fragment (kt, nt) = 32 lanes x 8 dwords, lane L holds column
// N = nt*16 + (L&15), K range (L>>4)*16 .. +15, VGPR v = K pair (2v, 2v+1).
// ---------------------------------------------------------------------------
__global__ void pack_w_kernel(const float* __restrict__ W, unsigned* __restrict__ Wp,
                              int Ktot, int Ntot) {
  int p = blockIdx.x * blockDim.x + threadIdx.x;
  int total = (Ktot * Ntot) >> 1;
  if (p >= total) return;
  int vv   = p & 7;
  int L    = (p >> 3) & 31;
  int tile = p >> 8;
  int ntiles = Ntot >> 4;
  int nt = tile % ntiles;
  int kt = tile / ntiles;
  int K = kt * 32 + (L >> 4) * 16 + 2 * vv;
  int N = nt * 16 + (L & 15);
  Wp[p] = bf16pack(W[(size_t)K * Ntot + N], W[(size_t)(K + 1) * Ntot + N]);
}

// ---------------------------------------------------------------------------
// C[M][256] = A[M][Ktot] (fp32, converted to bf16) @ Wp (packed bf16)
// Block: 256 threads = 8 waves. Tile: M=32, N=256. Wave w: mt = w>>2 (16-row
// slab), 4 n-tiles starting at (w&3)*4. A tile staged in LDS as fp32.
// B loads are issued at the top of each K-step so the barrier + LDS staging +
// bf16-pack VALU work hides their latency before the WMMA chain.
// ---------------------------------------------------------------------------
__launch_bounds__(256)
__global__ void gemm_bf16_n256(const float* __restrict__ A, int Ktot,
                               const unsigned* __restrict__ Wp,
                               float* __restrict__ C) {
  __shared__ float lds[32 * 36];  // 32 rows x 32 K, stride 36 dwords
  const int tid  = threadIdx.x;
  const int wave = tid >> 5;
  const int lane = tid & 31;
  const int mt   = wave >> 2;        // 0..1
  const int ntb  = (wave & 3) * 4;   // base n-tile (of 16)
  const int m0   = blockIdx.x * 32;

  v8f c0 = {}, c1 = {}, c2 = {}, c3 = {};

  const int mloc = mt * 16 + (lane & 15);
  const int hf   = lane >> 4;

  const int ksteps = Ktot >> 5;
  for (int kt = 0; kt < ksteps; ++kt) {
    // --- issue all B-fragment loads up front (latency hidden by staging) ---
    uint4 bq[8];
    {
      const unsigned* wbase = Wp + (size_t)kt * 16 * 256;  // 16 n-tiles * 256 dw
#pragma unroll
      for (int i = 0; i < 4; ++i) {
        const uint4* bp = (const uint4*)(wbase + (size_t)(ntb + i) * 256 + lane * 8);
        bq[2 * i]     = bp[0];
        bq[2 * i + 1] = bp[1];
      }
    }

    // --- stage A tile (fp32) into LDS ---
    __syncthreads();
    {
      int row = tid >> 3;
      int col = (tid & 7) * 4;
      float4 src = *(const float4*)(A + (size_t)(m0 + row) * Ktot + kt * 32 + col);
      *(float4*)(lds + row * 36 + col) = src;
    }
    __syncthreads();

    // --- build A fragment: 16-bit A 16x32 layout (lane = M, VGPR v -> K pair)
    union { v16bf v; unsigned u[8]; } af;
#pragma unroll
    for (int vv = 0; vv < 8; ++vv) {
      int k0 = hf * 16 + vv * 2;  // half*16+v*2
      float2 f = *(const float2*)(lds + mloc * 36 + k0);
      af.u[vv] = bf16pack(f.x, f.y);
    }

    // --- 4 WMMAs on the prefetched B fragments ---
#pragma unroll
    for (int i = 0; i < 4; ++i) {
      union { v16bf v; uint4 q[2]; } bf_;
      bf_.q[0] = bq[2 * i];
      bf_.q[1] = bq[2 * i + 1];
      v8f* cacc = (i == 0) ? &c0 : (i == 1) ? &c1 : (i == 2) ? &c2 : &c3;
      *cacc = __builtin_amdgcn_wmma_f32_16x16x32_bf16(
          false, af.v, false, bf_.v, (short)0, *cacc, false, false);
    }
  }

  // C/D layout: VGPR v, lanes0-15: M=v, N=lane; lanes16-31: M=v+8, N=lane-16
  const int mrow  = m0 + mt * 16 + hf * 8;
  const int ncol0 = lane & 15;
#pragma unroll
  for (int i = 0; i < 4; ++i) {
    v8f c = (i == 0) ? c0 : (i == 1) ? c1 : (i == 2) ? c2 : c3;
    float* dst = C + (size_t)mrow * 256 + (ntb + i) * 16 + ncol0;
#pragma unroll
    for (int vv = 0; vv < 8; ++vv) dst[(size_t)vv * 256] = c[vv];
  }
}

// ---------------------------------------------------------------------------
// SpMM: out[dst] += w * h[src], F1=256 features. One wave per edge; each lane
// handles 8 contiguous floats (2 x float4 load, 8 f32 atomic adds). Edge
// arrays are streamed once -> non-temporal loads keep L2 free for h/out rows.
// ---------------------------------------------------------------------------
__launch_bounds__(256)
__global__ void spmm_kernel(const int* __restrict__ esrc, const int* __restrict__ edst,
                            const float* __restrict__ ew, const float* __restrict__ h,
                            float* __restrict__ out) {
  int wid  = (blockIdx.x * blockDim.x + threadIdx.x) >> 5;
  int lane = threadIdx.x & 31;
  if (wid >= NE) return;
  int   s = __builtin_nontemporal_load(esrc + wid);
  int   d = __builtin_nontemporal_load(edst + wid);
  float w = __builtin_nontemporal_load(ew + wid);
  const float4* hp = (const float4*)(h + (size_t)s * F1) + lane * 2;
  float4 a = hp[0];
  float4 b = hp[1];
  float* op = out + (size_t)d * F1 + lane * 8;
  atomicAdd(op + 0, a.x * w);
  atomicAdd(op + 1, a.y * w);
  atomicAdd(op + 2, a.z * w);
  atomicAdd(op + 3, a.w * w);
  atomicAdd(op + 4, b.x * w);
  atomicAdd(op + 5, b.y * w);
  atomicAdd(op + 6, b.z * w);
  atomicAdd(op + 7, b.w * w);
}

__launch_bounds__(256)
__global__ void zero_kernel(float4* __restrict__ p, int n4) {
  int i = blockIdx.x * blockDim.x + threadIdx.x;
  if (i < n4) p[i] = make_float4(0.f, 0.f, 0.f, 0.f);
}

__launch_bounds__(256)
__global__ void elu_kernel(float* __restrict__ x, int n4) {
  int i = blockIdx.x * blockDim.x + threadIdx.x;
  if (i >= n4) return;
  float4 v = ((float4*)x)[i];
  v.x = v.x > 0.f ? v.x : expf(v.x) - 1.f;
  v.y = v.y > 0.f ? v.y : expf(v.y) - 1.f;
  v.z = v.z > 0.f ? v.z : expf(v.z) - 1.f;
  v.w = v.w > 0.f ? v.w : expf(v.w) - 1.f;
  ((float4*)x)[i] = v;
}

// ---------------------------------------------------------------------------
// Classifier: logits = z @ fc_w + fc_b, then row-wise log_softmax (40 classes).
// One 64-thread block per node row; z row cached in LDS.
// ---------------------------------------------------------------------------
__launch_bounds__(64)
__global__ void cls_kernel(const float* __restrict__ z, const float* __restrict__ fw,
                           const float* __restrict__ fb, float* __restrict__ out) {
  __shared__ float zr[F1];
  __shared__ float lg[NC];
  __shared__ float red[2];
  int r = blockIdx.x;
  int t = threadIdx.x;
  ((float4*)zr)[t] = ((const float4*)(z + (size_t)r * F1))[t];
  __syncthreads();
  if (t < NC) {
    float acc = fb[t];
#pragma unroll 4
    for (int k = 0; k < F1; ++k) acc += zr[k] * fw[k * NC + t];
    lg[t] = acc;
  }
  __syncthreads();
  if (t == 0) {
    float m = lg[0];
    for (int i = 1; i < NC; ++i) m = fmaxf(m, lg[i]);
    float s = 0.f;
    for (int i = 0; i < NC; ++i) s += expf(lg[i] - m);
    red[0] = m;
    red[1] = logf(s);
  }
  __syncthreads();
  if (t < NC) out[(size_t)r * NC + t] = lg[t] - red[0] - red[1];
}

extern "C" void kernel_launch(void* const* d_in, const int* in_sizes, int n_in,
                              void* d_out, int out_size, void* d_ws, size_t ws_size,
                              hipStream_t stream) {
  const float* x    = (const float*)d_in[0];
  const int*   esrc = (const int*)d_in[1];
  const int*   edst = (const int*)d_in[2];
  const float* ew   = (const float*)d_in[3];
  const float* W1   = (const float*)d_in[4];
  const float* W2   = (const float*)d_in[5];
  const float* fw   = (const float*)d_in[6];
  const float* fb   = (const float*)d_in[7];
  float* out = (float*)d_out;

  char* ws = (char*)d_ws;
  float*    bufA = (float*)ws;                                    // NN*F1 f32
  float*    bufB = (float*)(ws + (size_t)NN * F1 * 4);            // NN*F1 f32
  unsigned* w1p  = (unsigned*)(ws + (size_t)NN * F1 * 8);         // F0*F1 bf16
  unsigned* w2p  = (unsigned*)((char*)w1p + (size_t)F0 * F1 * 2); // F1*F1 bf16

  // pack weights to bf16 fragments
  {
    int p1 = (F0 * F1) / 2;  // 65536
    pack_w_kernel<<<p1 / 256, 256, 0, stream>>>(W1, w1p, F0, F1);
    int p2 = (F1 * F1) / 2;  // 32768
    pack_w_kernel<<<p2 / 256, 256, 0, stream>>>(W2, w2p, F1, F1);
  }

  const int n4     = (NN * F1) / 4;          // 6.4M float4s
  const int zgrid  = (n4 + 255) / 256;
  const int ggrid  = NN / 32;                // 3125 (exact)
  const int sgrid  = (NE * 32) / 256;        // 400000 (one wave per edge)

  // layer 1: H = X @ W1  -> bufA
  gemm_bf16_n256<<<ggrid, 256, 0, stream>>>(x, F0, w1p, bufA);
  // aggregate: bufB[dst] += w * bufA[src]
  zero_kernel<<<zgrid, 256, 0, stream>>>((float4*)bufB, n4);
  spmm_kernel<<<sgrid, 256, 0, stream>>>(esrc, edst, ew, bufA, bufB);
  elu_kernel<<<zgrid, 256, 0, stream>>>(bufB, n4);

  // layer 2: H2 = bufB @ W2 -> bufA
  gemm_bf16_n256<<<ggrid, 256, 0, stream>>>(bufB, F1, w2p, bufA);
  zero_kernel<<<zgrid, 256, 0, stream>>>((float4*)bufB, n4);
  spmm_kernel<<<sgrid, 256, 0, stream>>>(esrc, edst, ew, bufA, bufB);
  elu_kernel<<<zgrid, 256, 0, stream>>>(bufB, n4);

  // classifier + log_softmax
  cls_kernel<<<NN, 64, 0, stream>>>(bufB, fw, fb, out);
}
